// Masked_NG_VLToken_CA_53188874994190
// MI455X (gfx1250) — matrix operand order repaired and checked by reading, not compiled, not run on hardware
//
#include <hip/hip_runtime.h>
#include <hip/hip_bf16.h>

// ---------------------------------------------------------------------------
// Masked group-local VL cross-attention, CDNA5 (gfx1250), wave32 + WMMA bf16.
//   D=1024, H=16, HD=64.  mu/log_var/attn outputs concatenated into d_out.
// Pipeline: one-time fp32->bf16 conversion of activations/weights, then pure
// bf16 WMMA GEMMs (fp32 accum), group-local VALU attention, fused bf16
// epilogues for intermediates.
// ---------------------------------------------------------------------------

typedef __attribute__((ext_vector_type(16))) __bf16          v16bf;
typedef __attribute__((ext_vector_type(8)))  unsigned short  v8us;
typedef __attribute__((ext_vector_type(8)))  float           v8f;
typedef __attribute__((ext_vector_type(4)))  float           fx4;

// fp32 -> bf16, round-to-nearest-even, integer ops only
__device__ __forceinline__ unsigned short f2bf(float f) {
    unsigned u = __builtin_bit_cast(unsigned, f);
    u += 0x7FFFu + ((u >> 16) & 1u);
    return (unsigned short)(u >> 16);
}

struct bfpair { v8us lo, hi; };   // 32B, bit-identical to v16bf

// Load a 16-element bf16 fragment from two aligned 16B chunks.
__device__ __forceinline__ v16bf load_frag(const unsigned short* p0,
                                           const unsigned short* p1) {
    bfpair s{ *(const v8us*)p0, *(const v8us*)p1 };
    return __builtin_bit_cast(v16bf, s);
}

#define WMMA_BF16(A,B,C) \
    __builtin_amdgcn_wmma_f32_16x16x32_bf16(false,(A),false,(B),(short)0,(C),false,false)

// ---------------------------------------------------------------------------
// One-time fp32 -> bf16 conversion (8 elems / thread / iter, 16B stores).
// ---------------------------------------------------------------------------
__global__ void cvt_bf16_kernel(const float* __restrict__ in,
                                unsigned short* __restrict__ out, long long n)
{
    long long i = ((long long)blockIdx.x * blockDim.x + threadIdx.x) * 8;
    const long long stride = (long long)gridDim.x * blockDim.x * 8;
    for (; i < n; i += stride) {
        float4 a = *(const float4*)(in + i);
        float4 b = *(const float4*)(in + i + 4);
        v8us o;
        o[0]=f2bf(a.x); o[1]=f2bf(a.y); o[2]=f2bf(a.z); o[3]=f2bf(a.w);
        o[4]=f2bf(b.x); o[5]=f2bf(b.y); o[6]=f2bf(b.z); o[7]=f2bf(b.w);
        *(v8us*)(out + i) = o;
    }
}

// ---------------------------------------------------------------------------
// Y[M,N] = A[M,K] @ B[N,K]^T + bias[N], A/B bf16, fp32 accumulate.
// Block: 256 threads = 8 waves (4 M-waves x 2 N-waves) -> 128x128 tile.
// Wave: 32x64 via 2x4 of 16x16x32 WMMA tiles; K stepped by 32.
// Per-lane fragments (bf16 16x32 / 32x16 layouts):
//   A: row = m+(lane&15), chunks at K=(lane>>4)*8 (+0..7) and +16 (+16..23)
//   B: col = n+(lane&15), chunks at K=(lane>>4)*8 (+0..7) and +16 (+16..23)
// BF16OUT: write bf16 result (for intermediates consumed by later GEMMs).
// ---------------------------------------------------------------------------
template<bool BF16OUT>
__global__ __launch_bounds__(256)
void gemm_bf16_kernel(const unsigned short* __restrict__ A,
                      const unsigned short* __restrict__ B,
                      const float* __restrict__ bias,
                      float* __restrict__ Yf,
                      unsigned short* __restrict__ Yb,
                      int M, int N, int K)
{
    const int lane = threadIdx.x & 31;
    const int wave = threadIdx.x >> 5;
    const int m0 = blockIdx.x * 128 + (wave & 3) * 32;
    const int n0 = blockIdx.y * 128 + (wave >> 2) * 64;

    v8f c[2][4];
#pragma unroll
    for (int i = 0; i < 2; ++i)
#pragma unroll
        for (int j = 0; j < 4; ++j) c[i][j] = (v8f){};

    const int kc = (lane >> 4) * 8;    // A K-chunk base (0 or 8)
    const int kb = (lane >> 4) * 8;    // B K-chunk base (0 or 8)
    const unsigned short* pa[2];
    const unsigned short* pb[4];
#pragma unroll
    for (int i = 0; i < 2; ++i)
        pa[i] = A + (size_t)(m0 + i * 16 + (lane & 15)) * K + kc;
#pragma unroll
    for (int j = 0; j < 4; ++j)
        pb[j] = B + (size_t)(n0 + j * 16 + (lane & 15)) * K + kb;

    for (int k = 0; k < K; k += 32) {
        v16bf a[2], b[4];
#pragma unroll
        for (int i = 0; i < 2; ++i)
            a[i] = load_frag(pa[i] + k, pa[i] + k + 16);
#pragma unroll
        for (int j = 0; j < 4; ++j)
            b[j] = load_frag(pb[j] + k, pb[j] + k + 16);
#pragma unroll
        for (int i = 0; i < 2; ++i)
#pragma unroll
            for (int j = 0; j < 4; ++j)
                c[i][j] = WMMA_BF16(a[i], b[j], c[i][j]);
    }

    // C/D layout: VGPR r -> M = r (lanes 0-15) or r+8 (lanes 16-31); N = lane&15
    const int col  = lane & 15;
    const int rofs = (lane >> 4) * 8;
    float bv[4];
#pragma unroll
    for (int j = 0; j < 4; ++j) bv[j] = bias[n0 + j * 16 + col];

#pragma unroll
    for (int i = 0; i < 2; ++i) {
#pragma unroll
        for (int r = 0; r < 8; ++r) {
            const size_t row = (size_t)(m0 + i * 16 + rofs + r) * N;
#pragma unroll
            for (int j = 0; j < 4; ++j) {
                const float v = c[i][j][r] + bv[j];
                if (BF16OUT) Yb[row + n0 + j * 16 + col] = f2bf(v);
                else         Yf[row + n0 + j * 16 + col] = v;
            }
        }
    }
}

// ---------------------------------------------------------------------------
// Exclusive prefix sums of the split lists (G=32: trivial serial scan).
// ---------------------------------------------------------------------------
__global__ void prefix_kernel(const int* __restrict__ isplit, const int* __restrict__ tsplit,
                              int G, int* __restrict__ voff, int* __restrict__ loff)
{
    if (threadIdx.x == 0 && blockIdx.x == 0) {
        int a = 0, b = 0;
        for (int i = 0; i < G; ++i) {
            voff[i] = a; a += isplit[i];
            loff[i] = b; b += tsplit[i];
        }
        voff[G] = a; loff[G] = b;
    }
}

// ---------------------------------------------------------------------------
// Nontemporal zero fill for the (31/32-zero) 256MB attn output.
// ---------------------------------------------------------------------------
__global__ void zero_fill_kernel(fx4* __restrict__ p, long long n4)
{
    long long i = (long long)blockIdx.x * blockDim.x + threadIdx.x;
    const long long stride = (long long)gridDim.x * blockDim.x;
    fx4 z = {0.f, 0.f, 0.f, 0.f};
    for (; i < n4; i += stride) __builtin_nontemporal_store(z, p + i);
}

// ---------------------------------------------------------------------------
// Group-local attention. Block = (group g, head h), 128 threads (1 patch each,
// chunked if the group has >128 patches). K/V head slices staged in LDS,
// probs kept in padded LDS, ctx accumulated in regs, emitted as bf16 (its only
// consumer is the Wo GEMM). Probs written into the diagonal attn block.
// ---------------------------------------------------------------------------
#define MAXT 64
__global__ __launch_bounds__(128)
void attn_ctx_kernel(const float* __restrict__ Q, const float* __restrict__ Kp,
                     const float* __restrict__ Vp, const int* __restrict__ voff,
                     const int* __restrict__ loff, float* __restrict__ attn,
                     unsigned short* __restrict__ ctxbf, int SUMP, int BT, int Dm)
{
    const int g = blockIdx.x, h = blockIdx.y;
    const int p0 = voff[g], p1 = voff[g + 1];
    const int t0 = loff[g];
    int tlen = loff[g + 1] - t0;
    if (tlen > MAXT) tlen = MAXT;

    __shared__ float4 sK[MAXT * 16];
    __shared__ float4 sV[MAXT * 16];
    __shared__ float  sP[128][MAXT + 1];   // +1 pad: avoid 64-bank conflicts

    for (int i = threadIdx.x; i < tlen * 16; i += blockDim.x) {
        const int t = i >> 4, d4 = i & 15;
        sK[i] = ((const float4*)(Kp + (size_t)(t0 + t) * Dm + h * 64))[d4];
        sV[i] = ((const float4*)(Vp + (size_t)(t0 + t) * Dm + h * 64))[d4];
    }
    __syncthreads();

    for (int pb = p0; pb < p1; pb += 128) {
        const int p = pb + threadIdx.x;
        if (p < p1) {
            float4 q[16];
            const float4* qp = (const float4*)(Q + (size_t)p * Dm + h * 64);
#pragma unroll
            for (int i = 0; i < 16; ++i) q[i] = qp[i];

            float mx = -3.0e38f;
            for (int t = 0; t < tlen; ++t) {
                const float4* kr = &sK[t * 16];
                float s = 0.f;
#pragma unroll
                for (int i = 0; i < 16; ++i) {
                    float4 kv = kr[i];
                    s += q[i].x * kv.x + q[i].y * kv.y + q[i].z * kv.z + q[i].w * kv.w;
                }
                s *= 0.125f;                       // 1/sqrt(64)
                sP[threadIdx.x][t] = s;
                mx = fmaxf(mx, s);
            }
            float sum = 0.f;
            for (int t = 0; t < tlen; ++t) {
                float e = __expf(sP[threadIdx.x][t] - mx);
                sP[threadIdx.x][t] = e;
                sum += e;
            }
            const float inv = 1.0f / sum;

            float4 acc[16];
#pragma unroll
            for (int i = 0; i < 16; ++i) acc[i] = make_float4(0.f, 0.f, 0.f, 0.f);

            float* arow = attn + (size_t)h * SUMP * BT + (size_t)p * BT + t0;
            for (int t = 0; t < tlen; ++t) {
                const float pr = sP[threadIdx.x][t] * inv;
                __builtin_nontemporal_store(pr, arow + t);
                const float4* vr = &sV[t * 16];
#pragma unroll
                for (int i = 0; i < 16; ++i) {
                    float4 vv = vr[i];
                    acc[i].x += pr * vv.x; acc[i].y += pr * vv.y;
                    acc[i].z += pr * vv.z; acc[i].w += pr * vv.w;
                }
            }
            unsigned short* cp = ctxbf + (size_t)p * Dm + h * 64;
#pragma unroll
            for (int i = 0; i < 16; i += 2) {
                v8us o;
                o[0]=f2bf(acc[i].x);   o[1]=f2bf(acc[i].y);
                o[2]=f2bf(acc[i].z);   o[3]=f2bf(acc[i].w);
                o[4]=f2bf(acc[i+1].x); o[5]=f2bf(acc[i+1].y);
                o[6]=f2bf(acc[i+1].z); o[7]=f2bf(acc[i+1].w);
                *(v8us*)(cp + i * 4) = o;
            }
        }
    }
}

// ---------------------------------------------------------------------------
extern "C" void kernel_launch(void* const* d_in, const int* in_sizes, int n_in,
                              void* d_out, int out_size, void* d_ws, size_t ws_size,
                              hipStream_t stream)
{
    (void)n_in; (void)out_size; (void)ws_size;
    const int D = 1024, H = 16;
    const int SUMP = in_sizes[0] / D;   // 4096
    const int BT   = in_sizes[1] / D;   // 1024
    const int G    = in_sizes[2];       // 32

    const float* V_token = (const float*)d_in[0];
    const float* L_token = (const float*)d_in[1];
    const int*   isplit  = (const int*)d_in[2];
    const int*   tsplit  = (const int*)d_in[3];
    const float* Ww[6] = { (const float*)d_in[4],  (const float*)d_in[6],
                           (const float*)d_in[8],  (const float*)d_in[10],
                           (const float*)d_in[12], (const float*)d_in[14] };
    const float* Wb[6] = { (const float*)d_in[5],  (const float*)d_in[7],
                           (const float*)d_in[9],  (const float*)d_in[11],
                           (const float*)d_in[13], (const float*)d_in[15] };
    // order: Wq, Wk, Wv, Wo, Wm, Wv2

    // ---- workspace layout ----
    float* Q  = (float*)d_ws;                       // [SUMP, D] fp32
    float* Kp = Q  + (size_t)SUMP * D;              // [BT, D]
    float* Vp = Kp + (size_t)BT * D;                // [BT, D]
    unsigned short* Vtok_bf = (unsigned short*)(Vp + (size_t)BT * D); // [SUMP,D]
    unsigned short* Ltok_bf = Vtok_bf + (size_t)SUMP * D;             // [BT,D]
    unsigned short* Wbf[6];
    Wbf[0] = Ltok_bf + (size_t)BT * D;
    for (int i = 1; i < 6; ++i) Wbf[i] = Wbf[i-1] + (size_t)D * D;
    unsigned short* ctx_bf = Wbf[5] + (size_t)D * D;                  // [SUMP,D]
    unsigned short* out_bf = ctx_bf + (size_t)SUMP * D;               // [SUMP,D]
    int* voff = (int*)(out_bf + (size_t)SUMP * D);
    int* loff = voff + (G + 1);

    // ---- output layout: mu | log_var | attn ----
    float* mu     = (float*)d_out;
    float* logvar = mu + (size_t)SUMP * D;
    float* attn   = logvar + (size_t)SUMP * D;

    // 1) segment offsets
    prefix_kernel<<<1, 32, 0, stream>>>(isplit, tsplit, G, voff, loff);

    // 2) one-time fp32 -> bf16 conversions (bandwidth-trivial)
    cvt_bf16_kernel<<<1024, 256, 0, stream>>>(V_token, Vtok_bf, (long long)SUMP * D);
    cvt_bf16_kernel<<<512,  256, 0, stream>>>(L_token, Ltok_bf, (long long)BT * D);
    for (int i = 0; i < 6; ++i)
        cvt_bf16_kernel<<<512, 256, 0, stream>>>(Ww[i], Wbf[i], (long long)D * D);

    // 3) Q/K/V projections (bf16 WMMA, fp32 out for the attention stage)
    dim3 blk(256);
    dim3 grdQ(SUMP / 128, D / 128);
    dim3 grdL(BT / 128,   D / 128);
    gemm_bf16_kernel<false><<<grdQ, blk, 0, stream>>>(Vtok_bf, Wbf[0], Wb[0], Q,  nullptr, SUMP, D, D);
    gemm_bf16_kernel<false><<<grdL, blk, 0, stream>>>(Ltok_bf, Wbf[1], Wb[1], Kp, nullptr, BT,   D, D);
    gemm_bf16_kernel<false><<<grdL, blk, 0, stream>>>(Ltok_bf, Wbf[2], Wb[2], Vp, nullptr, BT,   D, D);

    // 4) zero-fill the attn tensor (nontemporal streaming stores)
    const long long n4 = (long long)H * SUMP * BT / 4;
    zero_fill_kernel<<<4096, 256, 0, stream>>>((fx4*)attn, n4);

    // 5) group-local attention: probs into diagonal blocks of attn, bf16 ctx
    attn_ctx_kernel<<<dim3(G, H), 128, 0, stream>>>(Q, Kp, Vp, voff, loff,
                                                    attn, ctx_bf, SUMP, BT, D);

    // 6) out = ctx @ Wo^T + b (bf16 epilogue: only consumed by next GEMMs)
    gemm_bf16_kernel<true><<<grdQ, blk, 0, stream>>>(ctx_bf, Wbf[3], Wb[3], nullptr, out_bf, SUMP, D, D);

    // 7) mu / log_var heads (fp32 into d_out)
    gemm_bf16_kernel<false><<<grdQ, blk, 0, stream>>>(out_bf, Wbf[4], Wb[4], mu,     nullptr, SUMP, D, D);
    gemm_bf16_kernel<false><<<grdQ, blk, 0, stream>>>(out_bf, Wbf[5], Wb[5], logvar, nullptr, SUMP, D, D);
}
